// MoERouter_22514218566415
// MI455X (gfx1250) — matrix operand order repaired and checked by reading, not compiled, not run on hardware
//
#include <hip/hip_runtime.h>
#include <hip/hip_bf16.h>
#include <math.h>

#define N_TOKENS 262144
#define D_MODEL  1024
#define N_EXPERTS 64

typedef __attribute__((ext_vector_type(16))) __bf16 v16bf;
typedef __attribute__((ext_vector_type(8)))  float  v8f;
typedef __attribute__((ext_vector_type(4)))  float  f4v;

// Workspace layout:
//   [0   .. 256)  bytes : float prob_sum[64]
//   [256 .. 512)  bytes : float imp_sum[64]   (accessed as acc[64..127])
//   [512 .. 512+262144) : bf16 packed weights, WMMA B-fragment order:
//       chunk(mat,kb,et) = 32 lanes x 16 bf16 (32B per lane, contiguous)

// ---------------------------------------------------------------------------
// Prep: fp32 weights -> bf16 in B-fragment order; zero aux accumulators.
// ---------------------------------------------------------------------------
__global__ __launch_bounds__(256) void prep_kernel(const float* __restrict__ wg,
                                                   const float* __restrict__ wn,
                                                   float* __restrict__ acc,
                                                   __bf16* __restrict__ wpack)
{
    int t = blockIdx.x * 256 + threadIdx.x;          // 131072 total
    if (t < 128) acc[t] = 0.0f;

    int mat = t >> 16;                               // 0 = gate, 1 = noise
    int rem = t & 0xFFFF;
    int e   = rem >> 10;                             // expert row
    int d   = rem & 1023;                            // k index
    float v = (mat ? wn : wg)[e * D_MODEL + d];

    int kb = d >> 5, kin = d & 31;
    int kh = (kin >> 3) & 1, j = (kin & 7) | ((kin >> 4) << 3); // B frag: lane half, pos
    int et = e >> 4,  n = e & 15;
    int lane = kh * 16 + n;
    size_t dst = ((size_t)((mat * 32 + kb) * 4 + et) * 32 + lane) * 16 + j;
    wpack[dst] = (__bf16)v;
}

// A fragment (16x32 bf16) from row-major fp32: lane m=lane&15, kh=lane>>4
__device__ __forceinline__ v16bf load_a_frag(const float* __restrict__ p0)
{
    f4v f0 = *(const f4v*)(p0);
    f4v f1 = *(const f4v*)(p0 + 4);
    f4v f2 = *(const f4v*)(p0 + 16);
    f4v f3 = *(const f4v*)(p0 + 20);
    v16bf a;
#pragma unroll
    for (int i = 0; i < 4; ++i) {
        a[i]      = (__bf16)f0[i];
        a[4 + i]  = (__bf16)f1[i];
        a[8 + i]  = (__bf16)f2[i];
        a[12 + i] = (__bf16)f3[i];
    }
    return a;
}

// ---------------------------------------------------------------------------
// Main: each wave = 32 tokens (two A tiles share every B load -> halves L2
// B traffic). Block = 2 waves = 64 tokens; post-phase: 1 thread per token.
// ---------------------------------------------------------------------------
__global__ __launch_bounds__(64) void router_kernel(const float* __restrict__ x,
                                                    const float* __restrict__ noise,
                                                    const __bf16* __restrict__ wpack,
                                                    float* __restrict__ gates,
                                                    int* __restrict__ idx_out,
                                                    float* __restrict__ aux_acc)
{
    __shared__ float lds[64 * 130];   // per token: [0..63]=clean, [64..127]=raw std
    __shared__ float iacc[64];        // block importance sums

    const int tid  = threadIdx.x;
    const int w    = tid >> 5;
    const int lane = tid & 31;
    const int row0 = blockIdx.x * 64;
    const int m    = lane & 15;       // A-frag row within tile
    const int kh   = lane >> 4;       // A/B-frag K-half select

    // ---------------- GEMM phase ----------------
    v8f ag0[4], ag1[4], an0[4], an1[4];
#pragma unroll
    for (int et = 0; et < 4; ++et)
#pragma unroll
        for (int j = 0; j < 8; ++j) {
            ag0[et][j] = 0.0f; ag1[et][j] = 0.0f;
            an0[et][j] = 0.0f; an1[et][j] = 0.0f;
        }

    const float*  xr0 = x + (size_t)(row0 + w * 32 + m) * D_MODEL;
    const float*  xr1 = xr0 + (size_t)16 * D_MODEL;
    const __bf16* wl  = wpack + lane * 16;

    for (int kb = 0; kb < 32; ++kb) {
        v16bf a0 = load_a_frag(xr0 + kb * 32 + kh * 8);
        v16bf a1 = load_a_frag(xr1 + kb * 32 + kh * 8);
#pragma unroll
        for (int et = 0; et < 4; ++et) {
            v16bf bg = *(const v16bf*)(wl + (size_t)((0 * 32 + kb) * 4 + et) * 512);
            ag0[et] = __builtin_amdgcn_wmma_f32_16x16x32_bf16(
                false, a0, false, bg, (short)0, ag0[et], false, false);
            ag1[et] = __builtin_amdgcn_wmma_f32_16x16x32_bf16(
                false, a1, false, bg, (short)0, ag1[et], false, false);
        }
#pragma unroll
        for (int et = 0; et < 4; ++et) {
            v16bf bn = *(const v16bf*)(wl + (size_t)((1 * 32 + kb) * 4 + et) * 512);
            an0[et] = __builtin_amdgcn_wmma_f32_16x16x32_bf16(
                false, a0, false, bn, (short)0, an0[et], false, false);
            an1[et] = __builtin_amdgcn_wmma_f32_16x16x32_bf16(
                false, a1, false, bn, (short)0, an1[et], false, false);
        }
    }

    // C layout: lane n=lane&15, VGPR j -> row (lane>>4)*8+j. Spill to LDS.
    {
        int mr0 = w * 32 + kh * 8;
        int mr1 = mr0 + 16;
        int col = lane & 15;
#pragma unroll
        for (int et = 0; et < 4; ++et)
#pragma unroll
            for (int j = 0; j < 8; ++j) {
                lds[(mr0 + j) * 130 + et * 16 + col]      = ag0[et][j];
                lds[(mr0 + j) * 130 + 64 + et * 16 + col] = an0[et][j];
                lds[(mr1 + j) * 130 + et * 16 + col]      = ag1[et][j];
                lds[(mr1 + j) * 130 + 64 + et * 16 + col] = an1[et][j];
            }
    }
    iacc[tid] = 0.0f;
    __syncthreads();

    // ---------------- per-token phase: 1 thread per token ----------------
    const int    tk    = tid;
    const int    token = row0 + tk;
    const float* nrow  = noise + (size_t)token * N_EXPERTS;
    float*       lrow  = &lds[tk * 130];

    float lgt[64];
    float v1 = -INFINITY, v2 = -INFINITY;
    int   i1 = 0, i2 = 0;
#pragma unroll
    for (int jj = 0; jj < 16; ++jj) {
        f4v nv = *(const f4v*)(nrow + jj * 4);
#pragma unroll
        for (int k = 0; k < 4; ++k) {
            int   e  = jj * 4 + k;
            float r  = lrow[64 + e];
            float sp = fmaxf(r, 0.0f) + log1pf(expf(-fabsf(r)));   // softplus
            float lg = lrow[e] + nv[k] * sp;
            lgt[e]   = lg;
            if (lg > v1)      { v2 = v1; i2 = i1; v1 = lg; i1 = e; }
            else if (lg > v2) { v2 = lg; i2 = e; }
        }
    }
    // top-2 softmax weights
    float e2  = expf(v2 - v1);
    float wt1 = 1.0f / (1.0f + e2);
    float wt2 = e2 * wt1;

    // full softmax denominator over 64 experts (max = v1)
    float s = 0.0f;
#pragma unroll
    for (int e = 0; e < 64; ++e) s += expf(lgt[e] - v1);
    float inv = 1.0f / s;

    // gates row (dense write -> no pre-zeroing needed)
    float* grow = gates + (size_t)token * N_EXPERTS;
#pragma unroll
    for (int jj = 0; jj < 16; ++jj) {
        f4v gv;
#pragma unroll
        for (int k = 0; k < 4; ++k) {
            int e = jj * 4 + k;
            gv[k] = (e == i1) ? wt1 : ((e == i2) ? wt2 : 0.0f);
        }
        *(f4v*)(grow + jj * 4) = gv;
    }
    // probabilities back into own (consumed) LDS row for block transpose-sum
#pragma unroll
    for (int e = 0; e < 64; ++e) lrow[e] = expf(lgt[e] - v1) * inv;

    atomicAdd(&iacc[i1], wt1);
    atomicAdd(&iacc[i2], wt2);
    idx_out[(size_t)token * 2]     = i1;
    idx_out[(size_t)token * 2 + 1] = i2;
    __syncthreads();

    // thread e: column-sum prob over the block's 64 tokens (conflict-free,
    // pitch 130 -> bank = (2*tk + e) % 64), then one global atomic per expert
    float ps = 0.0f;
#pragma unroll
    for (int t2 = 0; t2 < 64; ++t2) ps += lds[t2 * 130 + tid];
    atomicAdd(&aux_acc[tid],      ps);          // prob sums
    atomicAdd(&aux_acc[64 + tid], iacc[tid]);   // importance sums
}

// ---------------------------------------------------------------------------
// Aux loss: E * sum_e (imp[e]/N * prob[e]/N)
// ---------------------------------------------------------------------------
__global__ __launch_bounds__(64) void aux_kernel(const float* __restrict__ acc,
                                                 float* __restrict__ out_aux)
{
    __shared__ float red[64];
    int t = threadIdx.x;
    red[t] = acc[t] * acc[64 + t];
    __syncthreads();
    if (t == 0) {
        float s = 0.0f;
        for (int i = 0; i < 64; ++i) s += red[i];
        const float scale = 64.0f / ((float)N_TOKENS * (float)N_TOKENS);
        out_aux[0] = s * scale;
    }
}

// ---------------------------------------------------------------------------
extern "C" void kernel_launch(void* const* d_in, const int* in_sizes, int n_in,
                              void* d_out, int out_size, void* d_ws, size_t ws_size,
                              hipStream_t stream)
{
    const float* x     = (const float*)d_in[0];   // [N, D]
    const float* noise = (const float*)d_in[1];   // [N, E]
    const float* wg    = (const float*)d_in[2];   // [E, D]
    const float* wn    = (const float*)d_in[3];   // [E, D]

    float*  acc   = (float*)d_ws;                          // 128 floats
    __bf16* wpack = (__bf16*)((char*)d_ws + 512);          // 131072 bf16

    float* gates   = (float*)d_out;                                        // N*64
    int*   idx_out = (int*)((float*)d_out + (size_t)N_TOKENS * N_EXPERTS); // N*2
    float* aux_out = (float*)d_out + (size_t)N_TOKENS * N_EXPERTS
                   + (size_t)N_TOKENS * 2;                                 // 1

    prep_kernel<<<512, 256, 0, stream>>>(wg, wn, acc, wpack);
    router_kernel<<<N_TOKENS / 64, 64, 0, stream>>>(x, noise, wpack,
                                                    gates, idx_out, acc);
    aux_kernel<<<1, 64, 0, stream>>>(acc, aux_out);
}
